// PAttentionReadout_4140348473884
// MI455X (gfx1250) — matrix operand order repaired and checked by reading, not compiled, not run on hardware
//
#include <hip/hip_runtime.h>

// ---------------------------------------------------------------------------
// PAttentionReadout fused pipeline for MI455X (gfx1250, wave32, WMMA)
//
//   key = feat_i @ W_key            -> bf16 WMMA (v_wmma_f32_16x16x32_bf16)
//   e   = sigmoid(key + qry[seg]) @ W_e   (fused epilogue, LDS tile reduce)
//   segment softmax (sorted seg)    -> per-graph binary search + reductions
//   rst = segsum(alpha * feat_i)    -> streaming + f32 atomics
// ---------------------------------------------------------------------------

typedef __attribute__((ext_vector_type(16))) __bf16        bf16x16;
typedef __attribute__((ext_vector_type(8)))  float         f32x8;
typedef __attribute__((ext_vector_type(4)))  float         f32x4;
typedef __attribute__((ext_vector_type(4)))  unsigned int  u32x4;

#define DIM        128
#define NGRAPH     256
#define TILE_M     16     // nodes per tile
#define A_PITCH    136    // 128 + 8 bf16 pad -> conflict-free ds_load_b128
#define S_PITCH    132    // f32 sigmoid-tile pitch (pad vs 128)
#define SPLITS     16     // blocks per graph in the weighted-sum pass

__device__ __forceinline__ unsigned short f2bf(float f) {
    unsigned int u = __float_as_uint(f);
    unsigned int r = (u + 0x7FFFu + ((u >> 16) & 1u)) >> 16;   // RNE
    return (unsigned short)r;
}

__device__ __forceinline__ float fast_sigmoid(float x) {
    // v_exp_f32 + v_rcp_f32: plenty of precision for softmax logits
    return __builtin_amdgcn_rcpf(1.0f + __expf(-x));
}

// --- W_key -> per-lane bf16 B-fragment layout (contiguous k order, 16-bit B 32x16) --
// fragment storage: [ntile(8)][ktile(4)][lane(32)][16 bf16] contiguous.
// lane l holds column n = ntile*16 + (l&15); value i holds K = ktile*32 + (l>>4)*16 + i
__global__ void __launch_bounds__(256) wkey_swizzle_kernel(
        const float* __restrict__ W_key, unsigned short* __restrict__ Wswz) {
    int idx  = blockIdx.x * 256 + threadIdx.x;          // 0 .. 16383
    int i    = idx & 15;
    int lane = (idx >> 4) & 31;
    int kt   = (idx >> 9) & 3;
    int nt   = idx >> 11;
    int K    = kt * 32 + (lane >> 4) * 16 + i;
    int col  = nt * 16 + (lane & 15);
    Wswz[idx] = f2bf(W_key[K * DIM + col]);
}

// --- qry[b][d] = feat_u[b] @ W_user[:,d] + b_user[d]  (tiny, scalar FMA) ---
__global__ void __launch_bounds__(128) qry_kernel(
        const float* __restrict__ feat_u, const float* __restrict__ W_user,
        const float* __restrict__ b_user, float* __restrict__ qry) {
    int b = blockIdx.x, d = threadIdx.x;
    float acc = b_user[d];
#pragma unroll 8
    for (int k = 0; k < DIM; ++k)
        acc = fmaf(feat_u[b * DIM + k], W_user[k * DIM + d], acc);
    qry[b * DIM + d] = acc;
}

__global__ void __launch_bounds__(256) zero_out_kernel(float* __restrict__ out, int n) {
    int i = blockIdx.x * 256 + threadIdx.x;
    if (i < n) out[i] = 0.0f;
}

// --- main fused kernel: key GEMM (bf16 WMMA) + sigmoid + dot(W_e) -> e[N] --
__global__ void __launch_bounds__(256) fused_key_e_kernel(
        const float* __restrict__ feat_i, const int* __restrict__ seg,
        const float* __restrict__ qry, const unsigned short* __restrict__ Wswz,
        const float* __restrict__ W_e, float* __restrict__ e_out,
        int nNodes, int nTiles) {
    __shared__ __align__(16) unsigned short Abuf[TILE_M * A_PITCH];
    __shared__ __align__(16) float sTile[TILE_M][S_PITCH];
    __shared__ float eTile[TILE_M];
    __shared__ int   segTile[TILE_M];

    const int lane = threadIdx.x & 31;
    const int wave = threadIdx.x >> 5;        // this wave's 16 output cols
    const int h    = lane >> 4;               // lane half (0/1)
    const int ln   = lane & 15;

    // B fragments: loop-invariant, kept in VGPRs for the whole kernel
    bf16x16 bfrag[4];
#pragma unroll
    for (int kt = 0; kt < 4; ++kt) {
        const u32x4* p = (const u32x4*)(Wswz + ((wave * 4 + kt) * 32 + lane) * 16);
        union { u32x4 q[2]; bf16x16 v; } u;
        u.q[0] = p[0];
        u.q[1] = p[1];
        bfrag[kt] = u.v;
    }
    const float we = W_e[wave * 16 + ln];     // per-lane output-column weight

    for (int tile = blockIdx.x; tile < nTiles; tile += gridDim.x) {
        const long base = (long)tile * TILE_M;
        __syncthreads();                      // previous iter done with LDS

        // Stage A tile: 16 rows x 128 f32 -> bf16 LDS (coalesced b128 loads)
        {
            int t   = threadIdx.x;            // 256 threads x 8 elements
            int row = t >> 4;
            int k0  = (t & 15) * 8;
            long r  = base + row;
            if (r >= nNodes) r = nNodes - 1;  // tail clamp (N % 16 == 0 normally)
            const f32x4* src = (const f32x4*)(feat_i + r * DIM + k0);
            f32x4 v0 = src[0], v1 = src[1];
            u32x4 pk;
            pk.x = (unsigned)f2bf(v0.x) | ((unsigned)f2bf(v0.y) << 16);
            pk.y = (unsigned)f2bf(v0.z) | ((unsigned)f2bf(v0.w) << 16);
            pk.z = (unsigned)f2bf(v1.x) | ((unsigned)f2bf(v1.y) << 16);
            pk.w = (unsigned)f2bf(v1.z) | ((unsigned)f2bf(v1.w) << 16);
            *(u32x4*)(Abuf + row * A_PITCH + k0) = pk;
        }
        if (threadIdx.x < TILE_M) {
            eTile[threadIdx.x] = 0.0f;
            long r = base + threadIdx.x;
            segTile[threadIdx.x] = seg[r < nNodes ? r : (nNodes - 1)];
        }
        __syncthreads();

        // 4x v_wmma_f32_16x16x32_bf16 over K = 128
        // A 16x32 bf16 lane layout: lane row M = ln, value i -> K = kt*32 + h*16 + i
        f32x8 c = {};
#pragma unroll
        for (int kt = 0; kt < 4; ++kt) {
            const unsigned short* ap = Abuf + ln * A_PITCH + kt * 32 + h * 16;
            union { u32x4 q[2]; bf16x16 v; } ua;
            ua.q[0] = *(const u32x4*)(ap);        // K = b .. b+7
            ua.q[1] = *(const u32x4*)(ap + 8);    // K = b+8 .. b+15
            c = __builtin_amdgcn_wmma_f32_16x16x32_bf16(
                    false, ua.v, false, bfrag[kt], (short)0, c, false, false);
        }

        // Epilogue phase 1: c[r] = key[node = base + r + h*8][col = wave*16 + ln]
        // store sigmoid(key + qry) * W_e[col] into LDS tile (no dependent chains)
#pragma unroll
        for (int r = 0; r < 8; ++r) {
            int mrow = r + h * 8;
            int g    = segTile[mrow];
            float q  = qry[g * DIM + wave * 16 + ln];
            float s  = fast_sigmoid(c[r] + q);
            sTile[mrow][wave * 16 + ln] = s * we;
        }
        __syncthreads();

        // Epilogue phase 2: 256 threads -> (node, 8 contiguous cols) partials,
        // vector ds_load_b128 x2 + one shared ds_add_f32 each
        {
            int node = threadIdx.x >> 4;
            int c0   = (threadIdx.x & 15) * 8;
            const f32x4* p = (const f32x4*)&sTile[node][c0];
            f32x4 a = p[0], b2 = p[1];
            float partial = ((a.x + a.y) + (a.z + a.w)) +
                            ((b2.x + b2.y) + (b2.z + b2.w));
            atomicAdd(&eTile[node], partial);
        }
        __syncthreads();
        if (threadIdx.x < TILE_M) {
            long r = base + threadIdx.x;
            if (r < nNodes) e_out[r] = eTile[threadIdx.x];
        }
    }
}

// --- per-graph softmax stats: binary search sorted seg, then max & sum-exp --
__global__ void __launch_bounds__(256) seg_stats_kernel(
        const int* __restrict__ seg, const float* __restrict__ e,
        float* __restrict__ m_out, float* __restrict__ denom_out,
        int* __restrict__ start_out, int* __restrict__ end_out, int nNodes) {
    __shared__ int sRange[2];
    __shared__ float red[256];
    const int b = blockIdx.x;
    if (threadIdx.x < 2) {                    // lower_bound(b) / lower_bound(b+1)
        int tgt = b + threadIdx.x;
        int lo = 0, hi = nNodes;
        while (lo < hi) { int mid = (lo + hi) >> 1; if (seg[mid] < tgt) lo = mid + 1; else hi = mid; }
        sRange[threadIdx.x] = lo;
    }
    __syncthreads();
    const int s = sRange[0], en = sRange[1];
    if (threadIdx.x == 0) { start_out[b] = s; end_out[b] = en; }

    float mx = -__builtin_inff();
    for (int i = s + threadIdx.x; i < en; i += 256) mx = fmaxf(mx, e[i]);
    red[threadIdx.x] = mx; __syncthreads();
    for (int o = 128; o > 0; o >>= 1) {
        if (threadIdx.x < o) red[threadIdx.x] = fmaxf(red[threadIdx.x], red[threadIdx.x + o]);
        __syncthreads();
    }
    const float m = red[0];
    __syncthreads();

    float sum = 0.0f;
    for (int i = s + threadIdx.x; i < en; i += 256) sum += __expf(e[i] - m);
    red[threadIdx.x] = sum; __syncthreads();
    for (int o = 128; o > 0; o >>= 1) {
        if (threadIdx.x < o) red[threadIdx.x] += red[threadIdx.x + o];
        __syncthreads();
    }
    if (threadIdx.x == 0) { m_out[b] = m; denom_out[b] = red[0]; }
}

// --- rst[b] += alpha_i * feat_i[i], streaming feat_i coalesced -------------
__global__ void __launch_bounds__(128) weighted_sum_kernel(
        const float* __restrict__ feat_i, const float* __restrict__ e,
        const float* __restrict__ m, const float* __restrict__ denom,
        const int* __restrict__ start, const int* __restrict__ end,
        float* __restrict__ out) {
    const int b  = blockIdx.x / SPLITS;
    const int sp = blockIdx.x % SPLITS;
    const int s = start[b], en = end[b];
    const float mb = m[b];
    const float dn = denom[b];
    const float inv = dn > 0.0f ? __builtin_amdgcn_rcpf(dn) : 0.0f;
    const int d = threadIdx.x;
    float acc = 0.0f;
    for (int i = s + sp; i < en; i += SPLITS) {
        float alpha = __expf(e[i] - mb) * inv;
        acc = fmaf(alpha, feat_i[(long)i * DIM + d], acc);
    }
    atomicAdd(&out[b * DIM + d], acc);
}

// ---------------------------------------------------------------------------
extern "C" void kernel_launch(void* const* d_in, const int* in_sizes, int n_in,
                              void* d_out, int out_size, void* d_ws, size_t ws_size,
                              hipStream_t stream) {
    const float* feat_i = (const float*)d_in[0];
    const float* feat_u = (const float*)d_in[1];
    const int*   seg    = (const int*)d_in[2];
    const float* W_key  = (const float*)d_in[3];
    const float* W_user = (const float*)d_in[4];
    const float* b_user = (const float*)d_in[5];
    const float* W_e    = (const float*)d_in[6];
    float* out = (float*)d_out;

    const int nNodes = in_sizes[2];                // N (seg length)
    const int nTiles = (nNodes + TILE_M - 1) / TILE_M;

    // workspace layout (256B aligned chunks)
    char* ws = (char*)d_ws;
    size_t off = 0;
    auto take = [&](size_t bytes) { void* p = ws + off; off = (off + bytes + 255) & ~(size_t)255; return p; };
    float*          e_buf = (float*)take((size_t)nNodes * sizeof(float));
    float*          qry   = (float*)take((size_t)NGRAPH * DIM * sizeof(float));
    unsigned short* Wswz  = (unsigned short*)take(8 * 4 * 32 * 16 * sizeof(unsigned short));
    float*          m_b   = (float*)take(NGRAPH * sizeof(float));
    float*          d_b   = (float*)take(NGRAPH * sizeof(float));
    int*            s_b   = (int*)take(NGRAPH * sizeof(int));
    int*            en_b  = (int*)take(NGRAPH * sizeof(int));
    (void)ws_size;

    zero_out_kernel<<<(out_size + 255) / 256, 256, 0, stream>>>(out, out_size);
    wkey_swizzle_kernel<<<64, 256, 0, stream>>>(W_key, Wswz);
    qry_kernel<<<NGRAPH, DIM, 0, stream>>>(feat_u, W_user, b_user, qry);
    fused_key_e_kernel<<<2048, 256, 0, stream>>>(feat_i, seg, qry, Wswz, W_e,
                                                 e_buf, nNodes, nTiles);
    seg_stats_kernel<<<NGRAPH, 256, 0, stream>>>(seg, e_buf, m_b, d_b, s_b, en_b, nNodes);
    weighted_sum_kernel<<<NGRAPH * SPLITS, DIM, 0, stream>>>(feat_i, e_buf, m_b, d_b,
                                                             s_b, en_b, out);
}
